// MultiPhasePhasorModel_37082747633923
// MI455X (gfx1250) — matrix operand order repaired and checked by reading, not compiled, not run on hardware
//
#include <hip/hip_runtime.h>
#include <hip/hip_bf16.h>
#include <math.h>

typedef __attribute__((ext_vector_type(16))) __bf16 v16bf;
typedef __attribute__((ext_vector_type(8)))  __bf16 v8bf;
typedef __attribute__((ext_vector_type(4)))  __bf16 v4bf;
typedef __attribute__((ext_vector_type(8)))  float  v8f;
typedef __attribute__((ext_vector_type(4)))  float  v4f;
typedef __attribute__((ext_vector_type(4)))  unsigned int v4u;
typedef __attribute__((ext_vector_type(8)))  int    v8i;
typedef __attribute__((ext_vector_type(4)))  int    v4i;

#define D_MODEL 512
#define SEQ     2048
#define NBATCH  2
#define NROWS   4096   // B*L
#define NPH     8
#define NLAYER  4
#define VOCAB   32000

// ---------------- GEMM: C[M,N] = A_bf16[M,K] @ B_bf16[K,N] + bias (+res1 +res2) ----------------
// Block tile 128x256 (8 waves of 64x64), K-step 32, double-buffered LDS,
// A tile via Tensor Data Mover DMA, B tile via staged global loads + LDS transpose.
#define TILE_M 128
#define TILE_N 256
#define TILE_K 32
#define BSTR   40                    // padded K-stride (elements) of transposed B tile (80B rows)
#define SA_BYTES (TILE_M * TILE_K * 2)   // 8192
#define SB_BYTES (TILE_N * BSTR * 2)     // 20480
#define SH_BYTES (2 * SA_BYTES + 2 * SB_BYTES)  // 57344

__device__ __forceinline__ void tdm_load_a_tile(const __bf16* gsrc, unsigned ldsOff,
                                                int M, int K)
{
    unsigned long long gaddr = (unsigned long long)(size_t)gsrc;
    v4u g0;
    g0[0] = 1u;                                          // count=1, user descriptor
    g0[1] = ldsOff;                                      // LDS byte address of dest tile
    g0[2] = (unsigned)(gaddr & 0xFFFFFFFFu);
    g0[3] = (unsigned)((gaddr >> 32) & 0x1FFFFFFull) | (2u << 30); // type=2 ("image")
    v8i g1;
    g1[0] = 0x00010000;                                  // data_size=1 (2 bytes), no mask/pad
    g1[1] = (int)(((unsigned)K & 0xFFFFu) << 16);        // tensor_dim0 lo16
    g1[2] = (int)((((unsigned)K >> 16) & 0xFFFFu) |
                  (((unsigned)M & 0xFFFFu) << 16));      // dim0 hi16 | tensor_dim1 lo16
    g1[3] = (int)((((unsigned)M >> 16) & 0xFFFFu) |
                  ((unsigned)TILE_K << 16));             // dim1 hi16 | tile_dim0=32
    g1[4] = TILE_M;                                      // tile_dim1=128, tile_dim2=0
    g1[5] = K;                                           // tensor_dim0_stride lo32
    g1[6] = 0;
    g1[7] = 0;
    v4i gz4 = (v4i){0, 0, 0, 0};
    v8i gz8 = (v8i){0, 0, 0, 0, 0, 0, 0, 0};
    __builtin_amdgcn_tensor_load_to_lds(g0, g1, gz4, gz4, gz8, 0);
}

__global__ __launch_bounds__(256) void gemm_bf16_kernel(
    const __bf16* __restrict__ A, const __bf16* __restrict__ B,
    const float* __restrict__ bias, const float* __restrict__ res1,
    const float* __restrict__ res2, float* __restrict__ C,
    int M, int N, int K)
{
    extern __shared__ char smem[];   // [sA0][sA1][sB0][sB1]

    const int tid   = threadIdx.x;
    const int lane  = tid & 31;
    const int wid   = tid >> 5;          // 8 waves
    const int waveM = (wid >> 2) * 64;   // 2x4 wave grid of 64x64 tiles
    const int waveN = (wid & 3) * 64;
    const int mBase = blockIdx.x * TILE_M;
    const int nBase = blockIdx.y * TILE_N;
    const int nk    = K / TILE_K;

    v8f acc[4][4];
#pragma unroll
    for (int mi = 0; mi < 4; ++mi)
#pragma unroll
        for (int ni = 0; ni < 4; ++ni)
            acc[mi][ni] = (v8f){0.f,0.f,0.f,0.f,0.f,0.f,0.f,0.f};

    // B staging geometry: 32 K-rows x 256 N-cols, 4 passes of 8 rows
    const int bRow = tid >> 5;         // 0..7
    const int bCol = (tid & 31) * 8;   // 0..248

    // ---- prologue: tile 0 ----
    if (wid == 0)
        tdm_load_a_tile(A + (size_t)mBase * K, 0u, M, K);
    {
        v8bf st[4];
#pragma unroll
        for (int p = 0; p < 4; ++p)
            st[p] = *(const v8bf*)(B + (size_t)(bRow + p * 8) * N + nBase + bCol);
        __bf16* dB = (__bf16*)(smem + 2 * SA_BYTES);
#pragma unroll
        for (int p = 0; p < 4; ++p)
#pragma unroll
            for (int i = 0; i < 8; ++i)
                dB[(bCol + i) * BSTR + bRow + p * 8] = st[p][i];
    }
    if (wid == 0) __builtin_amdgcn_s_wait_tensorcnt(0);
    __syncthreads();

    const int arow  = lane & 15;
    const int koffA = (lane >> 4) * 8;   // A: lanes 0-15 K0..7/K16..23 ; 16-31 K8..15/K24..31
    const int koffB = (lane >> 4) * 8;   // B: lanes 0-15 K0..7/K16..23 ; 16-31 K8..15/K24..31

    for (int kt = 0; kt < nk; ++kt) {
        const int cur = kt & 1, nxt = cur ^ 1;
        const bool hasNext = (kt + 1) < nk;

        // ---- issue next tile's DMA + global loads before computing ----
        v8bf st[4];
        if (hasNext) {
            if (wid == 0)
                tdm_load_a_tile(A + (size_t)mBase * K + (kt + 1) * TILE_K,
                                (unsigned)(nxt * SA_BYTES), M, K);
            const __bf16* gB = B + (size_t)(kt + 1) * TILE_K * N + nBase + bCol;
#pragma unroll
            for (int p = 0; p < 4; ++p)
                st[p] = *(const v8bf*)(gB + (size_t)(bRow + p * 8) * N);
        }

        // ---- compute from current buffers ----
        const __bf16* sA = (const __bf16*)(smem + cur * SA_BYTES);
        const __bf16* sB = (const __bf16*)(smem + 2 * SA_BYTES + cur * SB_BYTES);

        v16bf bfr[4];
#pragma unroll
        for (int ni = 0; ni < 4; ++ni) {
            const __bf16* bp = sB + (size_t)(waveN + ni * 16 + (lane & 15)) * BSTR + koffB;
            union { v16bf v; v8bf h[2]; } u;
            u.h[0] = *(const v8bf*)(bp);
            u.h[1] = *(const v8bf*)(bp + 16);
            bfr[ni] = u.v;
        }
#pragma unroll
        for (int mi = 0; mi < 4; ++mi) {
            const __bf16* ap = sA + (size_t)(waveM + mi * 16 + arow) * TILE_K;
            union { v16bf v; v8bf h[2]; } u;
            u.h[0] = *(const v8bf*)(ap + koffA);
            u.h[1] = *(const v8bf*)(ap + 16 + koffA);
            v16bf afr = u.v;
#pragma unroll
            for (int ni = 0; ni < 4; ++ni)
                acc[mi][ni] = __builtin_amdgcn_wmma_f32_16x16x32_bf16(
                    false, afr, false, bfr[ni], (short)0, acc[mi][ni], false, false);
        }

        // ---- finish next-tile staging, wait DMA, single barrier ----
        if (hasNext) {
            __bf16* dB = (__bf16*)(smem + 2 * SA_BYTES + nxt * SB_BYTES);
#pragma unroll
            for (int p = 0; p < 4; ++p)
#pragma unroll
                for (int i = 0; i < 8; ++i)
                    dB[(bCol + i) * BSTR + bRow + p * 8] = st[p][i];
            if (wid == 0) __builtin_amdgcn_s_wait_tensorcnt(0);
        }
        __syncthreads();
    }

    // ---- store: C/D layout VGPR r -> M = r + (lane>=16)*8, N = lane&15 ----
    const int colIn   = lane & 15;
    const int rowHalf = (lane >> 4) * 8;
#pragma unroll
    for (int ni = 0; ni < 4; ++ni) {
        int col  = nBase + waveN + ni * 16 + colIn;
        float bb = bias ? bias[col] : 0.f;
#pragma unroll
        for (int mi = 0; mi < 4; ++mi) {
#pragma unroll
            for (int r = 0; r < 8; ++r) {
                int row = mBase + waveM + mi * 16 + rowHalf + r;
                size_t idx = (size_t)row * N + col;
                float v = acc[mi][ni][r] + bb;
                if (res1) v += res1[idx];
                if (res2) v += res2[idx];
                C[idx] = v;
            }
        }
    }
}

// ---------------- embedding gather ----------------
__global__ void gather_kernel(const int* __restrict__ tokens,
                              const float* __restrict__ embed,
                              float* __restrict__ h)
{
    int row = blockIdx.x;
    int t = tokens[row];
    const v4f* src = (const v4f*)(embed + (size_t)t * D_MODEL);
    v4f* dst = (v4f*)(h + (size_t)row * D_MODEL);
    dst[threadIdx.x] = src[threadIdx.x];   // blockDim = 128
}

// ---------------- layernorm (one wave per row), fp32 and/or bf16 out ----------------
__global__ __launch_bounds__(256) void ln_kernel(
    const float* __restrict__ x, const float* __restrict__ g,
    const float* __restrict__ b, float* __restrict__ o32,
    __bf16* __restrict__ obf, int rows)
{
    int wrow = (int)((blockIdx.x * blockDim.x + threadIdx.x) >> 5);
    int lane = threadIdx.x & 31;
    if (wrow >= rows) return;
    const v4f* xr = (const v4f*)(x + (size_t)wrow * D_MODEL);
    v4f v[4];
    float s = 0.f, s2 = 0.f;
#pragma unroll
    for (int i = 0; i < 4; ++i) {
        v[i] = xr[lane + i * 32];
#pragma unroll
        for (int j = 0; j < 4; ++j) { s += v[i][j]; s2 += v[i][j] * v[i][j]; }
    }
#pragma unroll
    for (int m = 16; m >= 1; m >>= 1) {
        s  += __shfl_xor(s,  m, 32);
        s2 += __shfl_xor(s2, m, 32);
    }
    float mean = s * (1.f / D_MODEL);
    float var  = s2 * (1.f / D_MODEL) - mean * mean;
    float rstd = rsqrtf(var + 1e-5f);
    const v4f* gv = (const v4f*)g;
    const v4f* bv = (const v4f*)b;
#pragma unroll
    for (int i = 0; i < 4; ++i) {
        int i4 = lane + i * 32;
        v4f gg = gv[i4], bb = bv[i4];
        v4f y;
#pragma unroll
        for (int j = 0; j < 4; ++j) y[j] = (v[i][j] - mean) * rstd * gg[j] + bb[j];
        if (o32) ((v4f*)(o32 + (size_t)wrow * D_MODEL))[i4] = y;
        if (obf) {
            v4bf yb;
#pragma unroll
            for (int j = 0; j < 4; ++j) yb[j] = (__bf16)y[j];
            ((v4bf*)(obf + (size_t)wrow * D_MODEL))[i4] = yb;
        }
    }
}

// ---------------- coef: per row, P=8 phase/amp projections -> scalar ----------------
__global__ __launch_bounds__(256) void coef_kernel(
    const float* __restrict__ hn, const float* __restrict__ pW,
    const float* __restrict__ pb, const float* __restrict__ aW,
    const float* __restrict__ ab, float* __restrict__ coef, int rows)
{
    int wrow = (int)((blockIdx.x * blockDim.x + threadIdx.x) >> 5);
    int lane = threadIdx.x & 31;
    if (wrow >= rows) return;
    const float* xr = hn + (size_t)wrow * D_MODEL;
    float pacc[NPH], aacc[NPH];
#pragma unroll
    for (int p = 0; p < NPH; ++p) { pacc[p] = 0.f; aacc[p] = 0.f; }
    for (int i = 0; i < 16; ++i) {
        int c = lane + i * 32;
        float e = xr[c];
        const float* pw = pW + (size_t)c * NPH;
        const float* aw = aW + (size_t)c * NPH;
#pragma unroll
        for (int p = 0; p < NPH; ++p) { pacc[p] += e * pw[p]; aacc[p] += e * aw[p]; }
    }
#pragma unroll
    for (int p = 0; p < NPH; ++p)
#pragma unroll
        for (int m = 16; m >= 1; m >>= 1) {
            pacc[p] += __shfl_xor(pacc[p], m, 32);
            aacc[p] += __shfl_xor(aacc[p], m, 32);
        }
    if (lane == 0) {
        float cf = 0.f;
#pragma unroll
        for (int p = 0; p < NPH; ++p) {
            float ph = tanhf(pacc[p] + pb[p]) * 3.14159265358979323846f;
            float t  = aacc[p] + ab[p];
            float sp = (t > 20.f) ? t : log1pf(expf(t));
            float am = sp + 0.1f;
            cf += am * (cosf(ph) + sinf(ph));
        }
        coef[wrow] = cf;
    }
}

// ---------------- blocked scan over L (32 chunks of 64) ----------------
__global__ __launch_bounds__(256) void scan_partial_kernel(
    const float* __restrict__ val, const float* __restrict__ coef,
    float* __restrict__ chunk)   // chunk[B][32][512]
{
    int b = blockIdx.y, ck = blockIdx.x;
    int ch = threadIdx.x * 2;
    float ax = 0.f, ay = 0.f;
    int l0 = ck * 64;
    for (int l = 0; l < 64; ++l) {
        int row = b * SEQ + l0 + l;
        float cf = coef[row];
        const float2 v = *(const float2*)(val + (size_t)row * D_MODEL + ch);
        ax += v.x * cf; ay += v.y * cf;
    }
    float2 o; o.x = ax; o.y = ay;
    *(float2*)(chunk + ((size_t)(b * 32 + ck)) * D_MODEL + ch) = o;
}

__global__ void scan_prefix_kernel(float* __restrict__ chunk)
{
    int idx = blockIdx.x * blockDim.x + threadIdx.x;   // 0..1023
    int b = idx >> 9, ch = idx & 511;
    float run = 0.f;
    for (int c = 0; c < 32; ++c) {
        float* p = chunk + ((size_t)(b * 32 + c)) * D_MODEL + ch;
        float t = *p;
        *p = run;
        run += t;
    }
}

__global__ __launch_bounds__(256) void scan_final_kernel(
    float* __restrict__ val, const float* __restrict__ coef,
    const float* __restrict__ chunk, float rs)
{
    int b = blockIdx.y, ck = blockIdx.x;
    int ch = threadIdx.x * 2;
    const float2 off = *(const float2*)(chunk + ((size_t)(b * 32 + ck)) * D_MODEL + ch);
    float ax = off.x, ay = off.y;
    int l0 = ck * 64;
    for (int l = 0; l < 64; ++l) {
        int row = b * SEQ + l0 + l;
        float cf = coef[row];
        float* p = val + (size_t)row * D_MODEL + ch;
        float2 v = *(const float2*)p;
        ax += v.x * cf; ay += v.y * cf;
        float2 o; o.x = ax * rs; o.y = ay * rs;
        *(float2*)p = o;                                 // retrieved, in place
    }
}

// ---------------- fp32 -> bf16 convert ----------------
__global__ void f2bf_kernel(const float* __restrict__ x, __bf16* __restrict__ y, long n)
{
    long i = ((long)blockIdx.x * blockDim.x + threadIdx.x) * 4;
    if (i < n) {
        v4f v = *(const v4f*)(x + i);
        v4bf o;
#pragma unroll
        for (int j = 0; j < 4; ++j) o[j] = (__bf16)v[j];
        *(v4bf*)(y + i) = o;
    }
}

// ---------------- orchestration ----------------
extern "C" void kernel_launch(void* const* d_in, const int* in_sizes, int n_in,
                              void* d_out, int out_size, void* d_ws, size_t ws_size,
                              hipStream_t stream) {
    (void)in_sizes; (void)n_in; (void)out_size; (void)ws_size;
    const int*   tokens        = (const int*)d_in[0];
    const float* embed         = (const float*)d_in[1];
    const float* ln_scale      = (const float*)d_in[2];
    const float* ln_bias       = (const float*)d_in[3];
    const float* phase_W       = (const float*)d_in[4];
    const float* phase_b       = (const float*)d_in[5];
    const float* amp_W         = (const float*)d_in[6];
    const float* amp_b         = (const float*)d_in[7];
    const float* value_W       = (const float*)d_in[8];
    const float* value_b       = (const float*)d_in[9];
    const float* out_ln_scale  = (const float*)d_in[10];
    const float* out_ln_bias   = (const float*)d_in[11];
    const float* out_W         = (const float*)d_in[12];
    const float* out_b         = (const float*)d_in[13];
    const float* normout_scale = (const float*)d_in[14];
    const float* normout_bias  = (const float*)d_in[15];
    const float* head_W        = (const float*)d_in[16];
    const float* head_b        = (const float*)d_in[17];
    float* out = (float*)d_out;

    char* ws = (char*)d_ws;
    float*  h      = (float*)(ws + 0);            // 8 MB
    float*  hn32   = (float*)(ws + 8388608);      // 8 MB
    __bf16* hnbf   = (__bf16*)(ws + 16777216);    // 4 MB
    float*  val    = (float*)(ws + 20971520);     // 8 MB (becomes retrieved in place)
    __bf16* rlnbf  = (__bf16*)(ws + 29360128);    // 4 MB
    float*  coefb  = (float*)(ws + 33554432);     // 16 KB
    float*  chunk  = (float*)(ws + 33570816);     // 128 KB
    __bf16* wbf    = (__bf16*)(ws + 33701888);    // 512 KB (per-layer weight)
    __bf16* headbf = (__bf16*)(ws + 34226176);    // 32.8 MB

    const float rs = 0.022097086912079608f;  // 1/sqrt(2048)

    // head_W fp32 -> bf16 (512*32000 = 16,384,000 elems)
    f2bf_kernel<<<16000, 256, 0, stream>>>(head_W, headbf, (long)D_MODEL * VOCAB);
    // h = embed[tokens]
    gather_kernel<<<NROWS, 128, 0, stream>>>(tokens, embed, h);

    for (int i = 0; i < NLAYER; ++i) {
        ln_kernel<<<512, 256, 0, stream>>>(h, ln_scale + i * D_MODEL, ln_bias + i * D_MODEL,
                                           hn32, hnbf, NROWS);
        coef_kernel<<<512, 256, 0, stream>>>(hn32,
                                             phase_W + (size_t)i * D_MODEL * NPH, phase_b + i * NPH,
                                             amp_W + (size_t)i * D_MODEL * NPH, amp_b + i * NPH,
                                             coefb, NROWS);
        f2bf_kernel<<<256, 256, 0, stream>>>(value_W + (size_t)i * D_MODEL * D_MODEL, wbf,
                                             (long)D_MODEL * D_MODEL);
        gemm_bf16_kernel<<<dim3(NROWS / TILE_M, D_MODEL / TILE_N), 256, SH_BYTES, stream>>>(
            hnbf, wbf, value_b + i * D_MODEL, nullptr, nullptr, val,
            NROWS, D_MODEL, D_MODEL);
        scan_partial_kernel<<<dim3(32, NBATCH), 256, 0, stream>>>(val, coefb, chunk);
        scan_prefix_kernel<<<4, 256, 0, stream>>>(chunk);
        scan_final_kernel<<<dim3(32, NBATCH), 256, 0, stream>>>(val, coefb, chunk, rs);
        ln_kernel<<<512, 256, 0, stream>>>(val, out_ln_scale + i * D_MODEL,
                                           out_ln_bias + i * D_MODEL, nullptr, rlnbf, NROWS);
        f2bf_kernel<<<256, 256, 0, stream>>>(out_W + (size_t)i * D_MODEL * D_MODEL, wbf,
                                             (long)D_MODEL * D_MODEL);
        // h = h + hn + LN(retrieved) @ out_W + out_b
        gemm_bf16_kernel<<<dim3(NROWS / TILE_M, D_MODEL / TILE_N), 256, SH_BYTES, stream>>>(
            rlnbf, wbf, out_b + i * D_MODEL, h, hn32, h,
            NROWS, D_MODEL, D_MODEL);
    }

    // final LN + head GEMM -> logits
    ln_kernel<<<512, 256, 0, stream>>>(h, normout_scale, normout_bias, nullptr, hnbf, NROWS);
    gemm_bf16_kernel<<<dim3(NROWS / TILE_M, VOCAB / TILE_N), 256, SH_BYTES, stream>>>(
        hnbf, headbf, head_b, nullptr, nullptr, out,
        NROWS, VOCAB, D_MODEL);
}
